// DecoderLSTMAttention_52278341927647
// MI455X (gfx1250) — compile-verified, matrix operand order checked by reading, not run on hardware
//
#include <hip/hip_runtime.h>
#include <math.h>

// ---------------------------------------------------------------------------
// bf16 helpers (raw u16 storage; manual round-to-nearest-even)
// ---------------------------------------------------------------------------
typedef __attribute__((ext_vector_type(16))) __bf16 v16bf;
typedef __attribute__((ext_vector_type(8)))  __bf16 v8bf;
typedef __attribute__((ext_vector_type(8)))  float  v8f;

__device__ __forceinline__ unsigned short f2bf(float f) {
    unsigned int u = __float_as_uint(f);
    u += 0x7FFFu + ((u >> 16) & 1u);
    return (unsigned short)(u >> 16);
}
__device__ __forceinline__ float bf2f(unsigned short h) {
    return __uint_as_float(((unsigned int)h) << 16);
}

// Problem constants
#define BB   64
#define NN   196
#define ENC  2048
#define DEC  512
#define ATT  512
#define EMBD 512
#define VV   10000
#define TT   20
#define KX   3072   // EMB + ENC + DEC (fused x|context|h)

// Load a 16x32 bf16 fragment row per CDNA5 ISA 7.12.2:
// lane&15 = row; lane>>4 selects K-half (kh = 0/8); VGPRs 0..3 = K[kh..kh+7],
// VGPRs 4..7 = K[16+kh..16+kh+7] -> two b128 loads.
__device__ __forceinline__ v16bf ldfrag(const unsigned short* __restrict__ row,
                                        int k0, int kh) {
    v8bf lo = *(const v8bf*)(row + k0 + kh);
    v8bf hi = *(const v8bf*)(row + k0 + 16 + kh);
    v16bf r;
#pragma unroll
    for (int i = 0; i < 8; ++i) { r[i] = lo[i]; r[8 + i] = hi[i]; }
    return r;
}

// ---------------------------------------------------------------------------
// bf16 WMMA GEMM: C[M,N](f32) = A[M,K](bf16,row-major) @ Bt[N,K](bf16,row-major)
//                 + bias[N]
// One wave computes a 16x64 C strip (4 N-tiles, shared A fragment -> 4x A
// reuse, 4 WMMAs per k-step hide s_wait latency). 8 waves / 256-thread block.
// Ragged N (e.g. 625 tiles for V=10000): last strip clamps B row pointers so
// the k-loop stays branch-free and wave-uniform (EXEC all-ones for WMMA);
// stores are guarded per sub-tile.
// ---------------------------------------------------------------------------
__global__ void gemm_bf16_wmma(const unsigned short* __restrict__ A, int lda,
                               const unsigned short* __restrict__ Bt, int ldb,
                               const float* __restrict__ bias,
                               float* __restrict__ C, long ldc,
                               int Mtiles, int Ntiles, int K) {
    int nstrips = (Ntiles + 3) >> 2;
    int wave  = blockIdx.x * (blockDim.x >> 5) + (threadIdx.x >> 5);
    int total = Mtiles * nstrips;
    if (wave >= total) return;                 // wave-uniform exit
    int mt = wave / nstrips;
    int ns = wave % nstrips;
    int lane = threadIdx.x & 31;
    int r15  = lane & 15;
    int kh   = (lane >> 4) << 3;               // 0 or 8

    int nt0 = ns * 4;
    int nrem = Ntiles - nt0;                   // 1..4 valid sub-tiles
    int n1 = (nt0 + 1 < Ntiles) ? nt0 + 1 : Ntiles - 1;
    int n2 = (nt0 + 2 < Ntiles) ? nt0 + 2 : Ntiles - 1;
    int n3 = (nt0 + 3 < Ntiles) ? nt0 + 3 : Ntiles - 1;

    const unsigned short* arow  = A  + (size_t)(mt * 16 + r15) * lda;
    const unsigned short* brow0 = Bt + (size_t)(nt0 * 16 + r15) * ldb;
    const unsigned short* brow1 = Bt + (size_t)(n1  * 16 + r15) * ldb;
    const unsigned short* brow2 = Bt + (size_t)(n2  * 16 + r15) * ldb;
    const unsigned short* brow3 = Bt + (size_t)(n3  * 16 + r15) * ldb;

    v8f acc0 = {}, acc1 = {}, acc2 = {}, acc3 = {};
    for (int k0 = 0; k0 < K; k0 += 32) {
        __builtin_prefetch(arow  + k0 + 128, 0, 3);
        __builtin_prefetch(brow0 + k0 + 128, 0, 3);
        __builtin_prefetch(brow1 + k0 + 128, 0, 3);
        __builtin_prefetch(brow2 + k0 + 128, 0, 3);
        __builtin_prefetch(brow3 + k0 + 128, 0, 3);
        v16bf a  = ldfrag(arow,  k0, kh);
        v16bf b0 = ldfrag(brow0, k0, kh);
        v16bf b1 = ldfrag(brow1, k0, kh);
        v16bf b2 = ldfrag(brow2, k0, kh);
        v16bf b3 = ldfrag(brow3, k0, kh);
        acc0 = __builtin_amdgcn_wmma_f32_16x16x32_bf16(false, a, false, b0,
                                                       (short)0, acc0, false, false);
        acc1 = __builtin_amdgcn_wmma_f32_16x16x32_bf16(false, a, false, b1,
                                                       (short)0, acc1, false, false);
        acc2 = __builtin_amdgcn_wmma_f32_16x16x32_bf16(false, a, false, b2,
                                                       (short)0, acc2, false, false);
        acc3 = __builtin_amdgcn_wmma_f32_16x16x32_bf16(false, a, false, b3,
                                                       (short)0, acc3, false, false);
    }

    // C layout: lane = N column (lane&15), VGPR r -> M = r + 8*(lane>=16)
    int mbase = mt * 16 + ((lane >> 4) << 3);
    float* crow = C + (size_t)mbase * ldc;
#define STORE_TILE(J, ACC)                                                    \
    if (J < nrem) {                                                           \
        int n = (nt0 + J) * 16 + r15;                                         \
        float bv = bias ? bias[n] : 0.0f;                                     \
        _Pragma("unroll")                                                     \
        for (int r = 0; r < 8; ++r)                                           \
            crow[(size_t)r * ldc + n] = ACC[r] + bv;                          \
    }
    STORE_TILE(0, acc0)
    STORE_TILE(1, acc1)
    STORE_TILE(2, acc2)
    STORE_TILE(3, acc3)
#undef STORE_TILE
}

// ---------------------------------------------------------------------------
// Setup kernels
// ---------------------------------------------------------------------------
// Wt[n*K + k] = bf16(W[k*N + n])
__global__ void transpose_f32_to_bf16(const float* __restrict__ W,
                                      unsigned short* __restrict__ Wt,
                                      int K, int N) {
    long i = (long)blockIdx.x * blockDim.x + threadIdx.x;
    long total = (long)K * N;
    if (i >= total) return;
    int  n = (int)(i % N);
    long k = i / N;
    Wt[(size_t)n * K + k] = f2bf(W[i]);
}

// Fused [W_ih ; W_hh] -> WxT[2048][3072] bf16, bias = b_ih + b_hh
__global__ void build_WxT(const float* __restrict__ W_ih, const float* __restrict__ W_hh,
                          const float* __restrict__ b_ih, const float* __restrict__ b_hh,
                          unsigned short* __restrict__ WxT, float* __restrict__ biasc) {
    long i = (long)blockIdx.x * blockDim.x + threadIdx.x;
    const long total = (long)KX * (4 * DEC);
    if (i >= total) return;
    int  n = (int)(i % (4 * DEC));
    long k = i / (4 * DEC);
    float v = (k < (EMBD + ENC)) ? W_ih[k * (4 * DEC) + n]
                                 : W_hh[(k - (EMBD + ENC)) * (4 * DEC) + n];
    WxT[(size_t)n * KX + k] = f2bf(v);
    if (k == 0) biasc[n] = b_ih[n] + b_hh[n];
}

__global__ void convert_f32_to_bf16(const float* __restrict__ src,
                                    unsigned short* __restrict__ dst, long n) {
    long i = (long)blockIdx.x * blockDim.x + threadIdx.x;
    if (i < n) dst[i] = f2bf(src[i]);
}

// mean over N=196 per (b, e)
__global__ void mean_feat_kernel(const float* __restrict__ f,
                                 unsigned short* __restrict__ meanb) {
    int i = blockIdx.x * blockDim.x + threadIdx.x;
    if (i >= BB * ENC) return;
    int b = i >> 11, e = i & (ENC - 1);
    const float* p = f + (size_t)b * NN * ENC + e;
    float s = 0.0f;
    for (int n = 0; n < NN; ++n) s += p[(size_t)n * ENC];
    meanb[i] = f2bf(s * (1.0f / (float)NN));
}

__global__ void init_state(const float* __restrict__ h0f, const float* __restrict__ c0f,
                           float* __restrict__ c, unsigned short* __restrict__ hb,
                           unsigned short* __restrict__ xh) {
    int i = blockIdx.x * blockDim.x + threadIdx.x;
    if (i >= BB * DEC) return;
    int b = i >> 9, d = i & (DEC - 1);
    c[i] = c0f[i];
    unsigned short h16 = f2bf(h0f[i]);
    hb[i] = h16;
    xh[(size_t)b * KX + EMBD + ENC + d] = h16;
}

// ---------------------------------------------------------------------------
// Per-step kernels
// ---------------------------------------------------------------------------
// scores[n] = Ab + sum_a Aw[a]*tanh(u_hs[b,n,a] + proj[b,a]); softmax over n.
__global__ void attn_softmax(const float* __restrict__ u_hs, const float* __restrict__ proj,
                             const float* __restrict__ Aw, const float* __restrict__ Ab,
                             float* __restrict__ alpha_ws, float* __restrict__ alpha_out,
                             int t) {
    __shared__ float sc[NN];
    __shared__ float sinv;
    int b = blockIdx.x;
    int wv = threadIdx.x >> 5, lane = threadIdx.x & 31;
    const float* pr = proj + (size_t)b * ATT;
    for (int n = wv; n < NN; n += 8) {
        const float* u = u_hs + ((size_t)b * NN + n) * ATT;
        float p = 0.0f;
        for (int a = lane; a < ATT; a += 32)
            p += Aw[a] * tanhf(u[a] + pr[a]);
        for (int o = 16; o; o >>= 1) p += __shfl_xor(p, o, 32);
        if (lane == 0) sc[n] = p + Ab[0];
    }
    __syncthreads();
    if (threadIdx.x == 0) {
        float m = sc[0];
        for (int n = 1; n < NN; ++n) m = fmaxf(m, sc[n]);
        float s = 0.0f;
        for (int n = 0; n < NN; ++n) { float e = __expf(sc[n] - m); sc[n] = e; s += e; }
        sinv = 1.0f / s;
    }
    __syncthreads();
    float inv = sinv;
    for (int n = threadIdx.x; n < NN; n += blockDim.x) {
        float a = sc[n] * inv;
        alpha_ws[(size_t)b * NN + n] = a;
        alpha_out[((size_t)b * (TT - 1) + t) * NN + n] = a;
    }
}

// xh[b] = [ emb(captions[b,t]) | context(b) | (h filled by cell kernel) ]
__global__ void context_emb(const unsigned short* __restrict__ featb,
                            const float* __restrict__ alpha,
                            const float* __restrict__ emb,
                            const int* __restrict__ captions,
                            int t, unsigned short* __restrict__ xh) {
    int b = blockIdx.x;
    unsigned short* row = xh + (size_t)b * KX;
    int tok = captions[b * TT + t];
    for (int j = threadIdx.x; j < EMBD; j += blockDim.x)
        row[j] = f2bf(emb[(size_t)tok * EMBD + j]);
    const float* al = alpha + (size_t)b * NN;
    const unsigned short* fb = featb + (size_t)b * NN * ENC;
    for (int e = threadIdx.x; e < ENC; e += blockDim.x) {
        float s = 0.0f;
        for (int n = 0; n < NN; ++n)
            s += al[n] * bf2f(fb[(size_t)n * ENC + e]);
        row[EMBD + e] = f2bf(s);
    }
}

__global__ void lstm_cell(const float* __restrict__ gates, float* __restrict__ c,
                          unsigned short* __restrict__ hb, unsigned short* __restrict__ xh) {
    int i = blockIdx.x * blockDim.x + threadIdx.x;
    if (i >= BB * DEC) return;
    int b = i >> 9, d = i & (DEC - 1);
    const float* g = gates + (size_t)b * (4 * DEC);
    float ig = g[d], fg = g[DEC + d], gg = g[2 * DEC + d], og = g[3 * DEC + d];
    float si = 1.0f / (1.0f + __expf(-ig));
    float sf = 1.0f / (1.0f + __expf(-fg));
    float so = 1.0f / (1.0f + __expf(-og));
    float cn = sf * c[i] + si * tanhf(gg);
    float hn = so * tanhf(cn);
    c[i] = cn;
    unsigned short h16 = f2bf(hn);
    hb[i] = h16;
    xh[(size_t)b * KX + EMBD + ENC + d] = h16;
}

// ---------------------------------------------------------------------------
// Host launch
// ---------------------------------------------------------------------------
static inline int gblk(long n, int thr) { return (int)((n + thr - 1) / thr); }

extern "C" void kernel_launch(void* const* d_in, const int* in_sizes, int n_in,
                              void* d_out, int out_size, void* d_ws, size_t ws_size,
                              hipStream_t stream) {
    (void)in_sizes; (void)n_in; (void)out_size; (void)ws_size;
    const float* features = (const float*)d_in[0];
    const int*   captions = (const int*)d_in[1];
    const float* emb      = (const float*)d_in[3];
    const float* Uw       = (const float*)d_in[4];
    const float* Ub       = (const float*)d_in[5];
    const float* Ww       = (const float*)d_in[6];
    const float* Wb       = (const float*)d_in[7];
    const float* Aw       = (const float*)d_in[8];
    const float* Ab       = (const float*)d_in[9];
    const float* ihw      = (const float*)d_in[10];
    const float* ihb      = (const float*)d_in[11];
    const float* icw      = (const float*)d_in[12];
    const float* icb      = (const float*)d_in[13];
    const float* W_ih     = (const float*)d_in[14];
    const float* b_ih     = (const float*)d_in[15];
    const float* W_hh     = (const float*)d_in[16];
    const float* b_hh     = (const float*)d_in[17];
    const float* fcn_w    = (const float*)d_in[18];
    const float* fcn_b    = (const float*)d_in[19];

    char* ws = (char*)d_ws;
    size_t off = 0;
    auto walloc = [&](size_t bytes) -> void* {
        off = (off + 255) & ~(size_t)255;
        void* p = ws + off; off += bytes; return p;
    };
    unsigned short* featb = (unsigned short*)walloc((size_t)BB * NN * ENC * 2);
    float*          u_hs  = (float*)         walloc((size_t)BB * NN * ATT * 4);
    unsigned short* UwT   = (unsigned short*)walloc((size_t)ATT * ENC * 2);
    unsigned short* WwT   = (unsigned short*)walloc((size_t)ATT * DEC * 2);
    unsigned short* ihwT  = (unsigned short*)walloc((size_t)DEC * ENC * 2);
    unsigned short* icwT  = (unsigned short*)walloc((size_t)DEC * ENC * 2);
    unsigned short* WxT   = (unsigned short*)walloc((size_t)(4 * DEC) * KX * 2);
    unsigned short* fcnT  = (unsigned short*)walloc((size_t)VV * DEC * 2);
    float*          biasc = (float*)         walloc((size_t)(4 * DEC) * 4);
    unsigned short* meanb = (unsigned short*)walloc((size_t)BB * ENC * 2);
    float*          h0f   = (float*)         walloc((size_t)BB * DEC * 4);
    float*          c0f   = (float*)         walloc((size_t)BB * DEC * 4);
    float*          proj  = (float*)         walloc((size_t)BB * ATT * 4);
    float*          alpha = (float*)         walloc((size_t)BB * NN * 4);
    unsigned short* xh    = (unsigned short*)walloc((size_t)BB * KX * 2);
    float*          gates = (float*)         walloc((size_t)BB * 4 * DEC * 4);
    unsigned short* hb    = (unsigned short*)walloc((size_t)BB * DEC * 2);
    float*          cst   = (float*)         walloc((size_t)BB * DEC * 4);

    const int thr = 256;
    auto gemm = [&](const unsigned short* A, int lda, const unsigned short* Bt, int ldb,
                    const float* bias, float* C, long ldc, int M, int N, int K) {
        int ntiles  = N / 16;
        int nstrips = (ntiles + 3) / 4;
        int waves   = (M / 16) * nstrips;
        int blocks  = (waves + 7) / 8;   // 8 waves per 256-thread block
        gemm_bf16_wmma<<<blocks, thr, 0, stream>>>(A, lda, Bt, ldb, bias, C, ldc,
                                                   M / 16, ntiles, K);
    };

    // ---- one-time setup -------------------------------------------------
    transpose_f32_to_bf16<<<gblk((long)ENC * ATT, thr), thr, 0, stream>>>(Uw, UwT, ENC, ATT);
    transpose_f32_to_bf16<<<gblk((long)DEC * ATT, thr), thr, 0, stream>>>(Ww, WwT, DEC, ATT);
    transpose_f32_to_bf16<<<gblk((long)ENC * DEC, thr), thr, 0, stream>>>(ihw, ihwT, ENC, DEC);
    transpose_f32_to_bf16<<<gblk((long)ENC * DEC, thr), thr, 0, stream>>>(icw, icwT, ENC, DEC);
    transpose_f32_to_bf16<<<gblk((long)DEC * VV, thr), thr, 0, stream>>>(fcn_w, fcnT, DEC, VV);
    build_WxT<<<gblk((long)KX * 4 * DEC, thr), thr, 0, stream>>>(W_ih, W_hh, b_ih, b_hh, WxT, biasc);
    convert_f32_to_bf16<<<gblk((long)BB * NN * ENC, thr), thr, 0, stream>>>(
        features, featb, (long)BB * NN * ENC);
    mean_feat_kernel<<<gblk(BB * ENC, thr), thr, 0, stream>>>(features, meanb);

    // u_hs = features @ Uw + Ub   [12544 x 512, K=2048]
    gemm(featb, ENC, UwT, ENC, Ub, u_hs, ATT, BB * NN, ATT, ENC);
    // h0/c0 = mean_feat @ {ihw,icw} + {ihb,icb}
    gemm(meanb, ENC, ihwT, ENC, ihb, h0f, DEC, BB, DEC, ENC);
    gemm(meanb, ENC, icwT, ENC, icb, c0f, DEC, BB, DEC, ENC);
    init_state<<<gblk(BB * DEC, thr), thr, 0, stream>>>(h0f, c0f, cst, hb, xh);

    float* preds    = (float*)d_out;                       // [B, T-1, V]
    float* alph_out = preds + (size_t)BB * (TT - 1) * VV;  // [B, T-1, N]

    // ---- recurrence -----------------------------------------------------
    for (int t = 0; t < TT - 1; ++t) {
        // proj = h @ Ww + Wb
        gemm(hb, DEC, WwT, DEC, Wb, proj, ATT, BB, ATT, DEC);
        attn_softmax<<<BB, thr, 0, stream>>>(u_hs, proj, Aw, Ab, alpha, alph_out, t);
        context_emb<<<BB, thr, 0, stream>>>(featb, alpha, emb, captions, t, xh);
        // gates = [emb|ctx|h] @ [W_ih;W_hh] + (b_ih+b_hh)
        gemm(xh, KX, WxT, KX, biasc, gates, 4 * DEC, BB, 4 * DEC, KX);
        lstm_cell<<<gblk(BB * DEC, thr), thr, 0, stream>>>(gates, cst, hb, xh);
        // preds[:, t, :] = h @ fcn_w + fcn_b   (ldc = (T-1)*V strides over b)
        gemm(hb, DEC, fcnT, DEC, fcn_b, preds + (size_t)t * VV,
             (long)(TT - 1) * VV, BB, VV, DEC);
    }
}
